// CrossAttnUGCA_15393162789577
// MI455X (gfx1250) — matrix operand order repaired and checked by reading, not compiled, not run on hardware
//
#include <hip/hip_runtime.h>
#include <hip/hip_bf16.h>

#define BB   32
#define LL   2048
#define DD   1024
#define HH   8
#define HD   128
#define KSEL 24
#define GH   256      // D/4 hidden units
#define GIN  4096     // 4*D gate input width

typedef __bf16 bf16_t;
typedef __attribute__((ext_vector_type(16))) __bf16 v16bf;
typedef __attribute__((ext_vector_type(8)))  float  v8f;

// Native f32 -> bf16 conversion: lowers to v_cvt_pk_bf16_f32 on gfx1250.
__device__ __forceinline__ bf16_t f2bf(float f) { return (bf16_t)f; }

// Truncating a flat pointer into the LDS aperture yields the 32-bit LDS byte
// address (ISA 10.2).
__device__ __forceinline__ unsigned lds_addr32(const void* p) {
  return (unsigned)(size_t)p;
}

// CDNA5 async memory->LDS copy (ASYNCcnt-tracked, bypasses VGPRs).
__device__ __forceinline__ void async_load_lds_b64(unsigned lds, const void* gaddr) {
  asm volatile("global_load_async_to_lds_b64 %0, %1, off"
               :: "v"(lds), "v"((unsigned long long)(size_t)gaddr) : "memory");
}
__device__ __forceinline__ void async_load_lds_b32(unsigned lds, const void* gaddr) {
  asm volatile("global_load_async_to_lds_b32 %0, %1, off"
               :: "v"(lds), "v"((unsigned long long)(size_t)gaddr) : "memory");
}
__device__ __forceinline__ void wait_async0() {
  asm volatile("s_wait_asynccnt 0" ::: "memory");
}

__device__ __forceinline__ void load_a_frag(const float* __restrict__ p, v16bf& A) {
  // elements 0..15 <- p[0..15]  (K = khalf*16 + 0..15)
  float4 x0 = *(const float4*)(p);
  float4 x1 = *(const float4*)(p + 4);
  float4 x2 = *(const float4*)(p + 8);
  float4 x3 = *(const float4*)(p + 12);
  A[0]=f2bf(x0.x); A[1]=f2bf(x0.y); A[2]=f2bf(x0.z); A[3]=f2bf(x0.w);
  A[4]=f2bf(x1.x); A[5]=f2bf(x1.y); A[6]=f2bf(x1.z); A[7]=f2bf(x1.w);
  A[8]=f2bf(x2.x); A[9]=f2bf(x2.y); A[10]=f2bf(x2.z); A[11]=f2bf(x2.w);
  A[12]=f2bf(x3.x); A[13]=f2bf(x3.y); A[14]=f2bf(x3.z); A[15]=f2bf(x3.w);
}

// ---------------- prep: convert gW1, Wk, Wv to bf16 (row-major) -------------
__global__ void k_prep(const float* __restrict__ gW1, const float* __restrict__ Wk,
                       const float* __restrict__ Wv, bf16_t* __restrict__ gW1bf,
                       bf16_t* __restrict__ Wkbf, bf16_t* __restrict__ Wvbf) {
  int t = blockIdx.x * blockDim.x + threadIdx.x;
  int base = t * 4;                       // 1024 blocks * 256 thr * 4 = 1M each
  #pragma unroll
  for (int i = 0; i < 4; i++) {
    gW1bf[base + i] = f2bf(gW1[base + i]);
    Wkbf [base + i] = f2bf(Wk [base + i]);
    Wvbf [base + i] = f2bf(Wv [base + i]);
  }
}

// ---------------- c0[b][n] = sum_d q[b,d] * gW1[n,d]  (segment-0 constant) --
__global__ void k_c0(const float* __restrict__ q_vec, const float* __restrict__ gW1,
                     float* __restrict__ c0) {
  __shared__ float qs[DD];
  int b = blockIdx.x, t = threadIdx.x;
  for (int i = t; i < DD; i += 256) qs[i] = q_vec[b * DD + i];
  __syncthreads();
  const float* row = gW1 + (size_t)t * GIN;   // t = hidden index n (0..255)
  float s = 0.f;
  for (int d = 0; d < DD; d++) s += qs[d] * row[d];
  c0[b * GH + t] = s;
}

// ---------------- qproj[b][n] = q[b] @ Wq[n,:] + bq[n] ----------------------
__global__ void k_qproj(const float* __restrict__ q_vec, const float* __restrict__ Wq,
                        const float* __restrict__ bq, float* __restrict__ qproj) {
  __shared__ float qs[DD];
  int b = blockIdx.x, t = threadIdx.x;
  for (int i = t; i < DD; i += 256) qs[i] = q_vec[b * DD + i];
  __syncthreads();
  #pragma unroll
  for (int r = 0; r < 4; r++) {
    int n = t + 256 * r;
    const float* row = Wq + (size_t)n * DD;
    float s = bq[n];
    for (int d = 0; d < DD; d++) s += qs[d] * row[d];
    qproj[b * DD + n] = s;
  }
}

// ---------------- gate MLP: WMMA bf16, 3 streamed segments ------------------
// kv tile staged via GLOBAL_LOAD_ASYNC_TO_LDS (ASYNCcnt), stride 34 floats so
// every b64 LDS destination is 8-byte aligned and fragment reads stay
// bank-conflict free (gcd-based: 34*m distinct mod 64 for m in [0,16)).
__global__ void __launch_bounds__(256) k_gate(
    const float* __restrict__ q_vec, const float* __restrict__ kv,
    const bf16_t* __restrict__ gW1bf, const float* __restrict__ c0,
    const float* __restrict__ gb1, const float* __restrict__ gW2,
    const float* __restrict__ gb2,
    float* __restrict__ e_out, float* __restrict__ g_out) {
  __shared__ float kv_lds[16 * 34];
  __shared__ float q_lds[32];
  __shared__ float h_lds[16 * 257];
  __shared__ float gw2_lds[GH];

  int b  = blockIdx.x >> 7;        // 128 l-tiles per batch
  int lt = blockIdx.x & 127;
  int t  = threadIdx.x;
  int w  = t >> 5;                 // wave id (wave32)
  int ln = t & 31;
  int m_a   = ln & 15;             // A-matrix row held by this lane
  int khalf = ln >> 4;             // A/B K-half selector
  int ncol  = ln & 15;             // B/C column held by this lane

  if (t < GH) gw2_lds[t] = gW2[t];

  const float* kvbase = kv + ((size_t)b * LL + lt * 16) * DD;
  int srow = t >> 4;               // staging: token row 0..15
  int scol = (t & 15) * 2;         // staging: 2 consecutive floats (8B aligned)
  unsigned kv_dst = lds_addr32(&kv_lds[srow * 34 + scol]);
  unsigned q_dst  = lds_addr32(&q_lds[t & 31]);

  v8f acc0 = {}; v8f acc1 = {};

  for (int k0 = 0; k0 < DD; k0 += 32) {
    __syncthreads();               // previous tile's readers are done
    async_load_lds_b64(kv_dst, kvbase + (size_t)srow * DD + k0 + scol);
    if (t < 32)                    // wave 0 stages the q chunk
      async_load_lds_b32(q_dst, q_vec + b * DD + k0 + t);
    if (k0 + 32 < DD)
      __builtin_prefetch(kvbase + (size_t)srow * DD + k0 + 32 + scol, 0, 1);
    wait_async0();                 // this wave's async copies landed in LDS
    __syncthreads();               // all waves' copies visible

    // Build A fragments for the 3 streamed feature segments
    v16bf a1, a2, a3;
    #pragma unroll
    for (int i = 0; i < 16; i++) {
      int k = khalf * 16 + i;
      float kvv = kv_lds[m_a * 34 + k];
      float qq  = q_lds[k];
      a1[i] = f2bf(kvv);                 // segment 1: kv
      a2[i] = f2bf(fabsf(qq - kvv));     // segment 2: |q - kv|
      a3[i] = f2bf(qq * kvv);            // segment 3: q * kv
    }

    #pragma unroll
    for (int nt = 0; nt < 2; nt++) {
      int n = (2 * w + nt) * 16 + ncol;
      const bf16_t* brow = gW1bf + (size_t)n * GIN + k0 + khalf * 16;
      v16bf b1 = *(const v16bf*)(brow + 1024);
      v16bf b2 = *(const v16bf*)(brow + 2048);
      v16bf b3 = *(const v16bf*)(brow + 3072);
      v8f& acc = nt ? acc1 : acc0;
      acc = __builtin_amdgcn_wmma_f32_16x16x32_bf16(false, a1, false, b1, (short)0, acc, false, false);
      acc = __builtin_amdgcn_wmma_f32_16x16x32_bf16(false, a2, false, b2, (short)0, acc, false, false);
      acc = __builtin_amdgcn_wmma_f32_16x16x32_bf16(false, a3, false, b3, (short)0, acc, false, false);
    }
  }

  __syncthreads();
  #pragma unroll
  for (int nt = 0; nt < 2; nt++) {
    int n = (2 * w + nt) * 16 + ncol;
    float bias = c0[b * GH + n] + gb1[n];
    v8f acc = nt ? acc1 : acc0;
    #pragma unroll
    for (int r = 0; r < 8; r++) {
      int m = r + khalf * 8;             // C layout: lanes 16-31 hold M+8
      float z = acc[r] + bias;
      h_lds[m * 257 + n] = z > 0.f ? z : 0.f;
    }
  }
  __syncthreads();

  if (t < 16) {
    int m = t;
    float s = 0.f;
    for (int n = 0; n < GH; n++) s += h_lds[m * 257 + n] * gw2_lds[n];
    float z = s + gb2[0];
    float e = (z > 20.f) ? z : log1pf(__expf(z));   // softplus
    int tok = lt * 16 + m;
    e_out[b * LL + tok] = e;
    g_out[b * LL + tok] = e / (e + 1.f);            // g = 1 - 1/(e+1)
  }
}

// ---------------- top-24 selection (matches lax.top_k tie-breaking) ---------
__global__ void k_topk(const float* __restrict__ g_in, const unsigned char* __restrict__ mask,
                       int* __restrict__ topk, float* __restrict__ usel,
                       float* __restrict__ uavg) {
  __shared__ float sg[LL];
  __shared__ float rv[256];
  __shared__ int   ri[256];
  __shared__ int   sidx[KSEL];
  __shared__ float sgv[KSEL];
  int b = blockIdx.x, t = threadIdx.x;
  for (int i = t; i < LL; i += 256) {
    float g = g_in[b * LL + i];
    sg[i] = mask[b * LL + i] ? g : -INFINITY;
  }
  __syncthreads();
  for (int kk = 0; kk < KSEL; kk++) {
    float bvv = -INFINITY; int bii = 0x7fffffff;
    for (int i = t; i < LL; i += 256) {
      float v = sg[i];
      if (v > bvv) { bvv = v; bii = i; }
    }
    rv[t] = bvv; ri[t] = bii;
    __syncthreads();
    for (int s = 128; s > 0; s >>= 1) {
      if (t < s) {
        float v2 = rv[t + s]; int i2 = ri[t + s];
        if (v2 > rv[t] || (v2 == rv[t] && i2 < ri[t])) { rv[t] = v2; ri[t] = i2; }
      }
      __syncthreads();
    }
    if (t == 0) { sidx[kk] = ri[0]; sgv[kk] = rv[0]; sg[ri[0]] = -INFINITY; }
    __syncthreads();
  }
  if (t < KSEL) {
    topk[b * KSEL + t] = sidx[t];
    usel[b * KSEL + t] = 1.f - sgv[t];
  }
  if (t == 0) {
    float s = 0.f;
    for (int j = 0; j < KSEL; j++) s += 1.f - sgv[j];
    uavg[b] = s / (float)KSEL;
  }
}

// ---------------- gathered k/v projection: WMMA bf16 ------------------------
__global__ void __launch_bounds__(256) k_kvproj(
    const float* __restrict__ kv, const int* __restrict__ topk,
    const bf16_t* __restrict__ Wkbf, const bf16_t* __restrict__ Wvbf,
    const float* __restrict__ bk, const float* __restrict__ bvb,
    float* __restrict__ kproj, float* __restrict__ vproj) {
  __shared__ int sidx[KSEL];
  int b = blockIdx.x, t = threadIdx.x;
  int w = t >> 5, ln = t & 31;
  int khalf = ln >> 4, mloc = ln & 15, ncol = ln & 15;
  if (t < KSEL) sidx[t] = topk[b * KSEL + t];
  __syncthreads();

  int m1 = 16 + mloc;
  bool v1ok = (m1 < KSEL);
  const float* p0 = kv + ((size_t)b * LL + sidx[mloc]) * DD;
  const float* p1 = v1ok ? (kv + ((size_t)b * LL + sidx[m1]) * DD) : p0;

  for (int ntp = 0; ntp < 4; ntp++) {       // 8 waves * 4 pairs * 2 = 64 n-tiles
    int ntbase = w * 8 + ntp * 2;
    v8f accK[2][2]; v8f accV[2][2];
    #pragma unroll
    for (int j = 0; j < 2; j++) { accK[j][0] = (v8f){}; accK[j][1] = (v8f){};
                                  accV[j][0] = (v8f){}; accV[j][1] = (v8f){}; }
    for (int k0 = 0; k0 < DD; k0 += 32) {
      v16bf A0, A1;
      load_a_frag(p0 + k0 + khalf * 16, A0);
      if (v1ok) {
        load_a_frag(p1 + k0 + khalf * 16, A1);
      } else {
        bf16_t z = f2bf(0.f);
        #pragma unroll
        for (int i = 0; i < 16; i++) A1[i] = z;   // padded rows 24..31
      }
      #pragma unroll
      for (int j = 0; j < 2; j++) {
        int n = (ntbase + j) * 16 + ncol;
        v16bf Bk = *(const v16bf*)(Wkbf + (size_t)n * DD + k0 + khalf * 16);
        v16bf Bv = *(const v16bf*)(Wvbf + (size_t)n * DD + k0 + khalf * 16);
        accK[j][0] = __builtin_amdgcn_wmma_f32_16x16x32_bf16(false, A0, false, Bk, (short)0, accK[j][0], false, false);
        accK[j][1] = __builtin_amdgcn_wmma_f32_16x16x32_bf16(false, A1, false, Bk, (short)0, accK[j][1], false, false);
        accV[j][0] = __builtin_amdgcn_wmma_f32_16x16x32_bf16(false, A0, false, Bv, (short)0, accV[j][0], false, false);
        accV[j][1] = __builtin_amdgcn_wmma_f32_16x16x32_bf16(false, A1, false, Bv, (short)0, accV[j][1], false, false);
      }
    }
    #pragma unroll
    for (int j = 0; j < 2; j++) {
      int n = (ntbase + j) * 16 + ncol;
      float bkn = bk[n], bvn = bvb[n];
      #pragma unroll
      for (int r = 0; r < 8; r++) {
        int m = r + khalf * 8;                        // m-tile 0: rows 0..15
        kproj[((size_t)b * KSEL + m) * DD + n] = accK[j][0][r] + bkn;
        vproj[((size_t)b * KSEL + m) * DD + n] = accV[j][0][r] + bvn;
        int m2 = 16 + r + khalf * 8;                  // m-tile 1: rows 16..31
        if (m2 < KSEL) {
          kproj[((size_t)b * KSEL + m2) * DD + n] = accK[j][1][r] + bkn;
          vproj[((size_t)b * KSEL + m2) * DD + n] = accV[j][1][r] + bvn;
        }
      }
    }
  }
}

// ---------------- attention + o-proj + residual + LayerNorm -----------------
__global__ void k_attn(const float* __restrict__ qproj, const float* __restrict__ kproj,
                       const float* __restrict__ vproj, const float* __restrict__ usel,
                       const float* __restrict__ q_vec, const float* __restrict__ Wo,
                       const float* __restrict__ bo, const float* __restrict__ ln_g,
                       const float* __restrict__ ln_b,
                       float* __restrict__ fused, float* __restrict__ attn_out) {
  __shared__ float qs[DD], ol[DD], xl[DD];
  __shared__ float lg[HH * KSEL], wgt[HH * KSEL], ul[KSEL];
  __shared__ float red[256], red2[256];
  int b = blockIdx.x, t = threadIdx.x;
  for (int i = t; i < DD; i += 256) qs[i] = qproj[b * DD + i];
  if (t < KSEL) ul[t] = usel[b * KSEL + t];
  __syncthreads();

  if (t < HH * KSEL) {
    int h = t / KSEL, j = t % KSEL;
    const float* kr = kproj + ((size_t)b * KSEL + j) * DD + h * HD;
    const float* qr = qs + h * HD;
    float s = 0.f;
    for (int d = 0; d < HD; d++) s += qr[d] * kr[d];
    lg[t] = s * 0.08838834764831845f / (1.f + ul[j]);   // scale / temp
  }
  __syncthreads();
  if (t < HH) {
    float mx = -INFINITY;
    for (int j = 0; j < KSEL; j++) mx = fmaxf(mx, lg[t * KSEL + j]);
    float e[KSEL]; float sum = 0.f;
    for (int j = 0; j < KSEL; j++) { e[j] = __expf(lg[t * KSEL + j] - mx); sum += e[j]; }
    for (int j = 0; j < KSEL; j++) {
      float wv = e[j] / sum;
      wgt[t * KSEL + j] = wv;
      attn_out[b * HH * KSEL + t * KSEL + j] = wv;
    }
  }
  __syncthreads();
  #pragma unroll
  for (int r = 0; r < 4; r++) {
    int dd = t + 256 * r;
    int h = dd >> 7;
    float s = 0.f;
    for (int j = 0; j < KSEL; j++)
      s += wgt[h * KSEL + j] * vproj[((size_t)b * KSEL + j) * DD + dd];
    ol[dd] = s;
  }
  __syncthreads();
  float ls = 0.f, ls2 = 0.f;
  #pragma unroll
  for (int r = 0; r < 4; r++) {
    int n = t + 256 * r;
    const float* row = Wo + (size_t)n * DD;
    float s = bo[n];
    for (int d = 0; d < DD; d++) s += ol[d] * row[d];
    float x = q_vec[b * DD + n] + s;   // residual
    xl[n] = x; ls += x; ls2 += x * x;
  }
  red[t] = ls; red2[t] = ls2;
  __syncthreads();
  for (int s = 128; s > 0; s >>= 1) {
    if (t < s) { red[t] += red[t + s]; red2[t] += red2[t + s]; }
    __syncthreads();
  }
  float mu  = red[0] / (float)DD;
  float var = red2[0] / (float)DD - mu * mu;
  float inv = rsqrtf(var + 1e-5f);
  #pragma unroll
  for (int r = 0; r < 4; r++) {
    int n = t + 256 * r;
    fused[b * DD + n] = (xl[n] - mu) * inv * ln_g[n] + ln_b[n];
  }
}

// ---------------------------------------------------------------------------
extern "C" void kernel_launch(void* const* d_in, const int* in_sizes, int n_in,
                              void* d_out, int out_size, void* d_ws, size_t ws_size,
                              hipStream_t stream) {
  const float* q_vec = (const float*)d_in[0];
  const float* kv    = (const float*)d_in[1];
  const unsigned char* mask = (const unsigned char*)d_in[2];
  const float* Wq = (const float*)d_in[3];  const float* bq = (const float*)d_in[4];
  const float* Wk = (const float*)d_in[5];  const float* bk = (const float*)d_in[6];
  const float* Wv = (const float*)d_in[7];  const float* bv = (const float*)d_in[8];
  const float* Wo = (const float*)d_in[9];  const float* bo = (const float*)d_in[10];
  const float* gW1 = (const float*)d_in[11]; const float* gb1 = (const float*)d_in[12];
  const float* gW2 = (const float*)d_in[13]; const float* gb2 = (const float*)d_in[14];
  const float* ln_g = (const float*)d_in[15]; const float* ln_b = (const float*)d_in[16];

  char* ws = (char*)d_ws;
  bf16_t* gW1bf = (bf16_t*)(ws + 0);          // 2 MB
  bf16_t* Wkbf  = (bf16_t*)(ws + 2097152);    // 2 MB
  bf16_t* Wvbf  = (bf16_t*)(ws + 4194304);    // 2 MB
  float*  c0    = (float*) (ws + 6291456);    // 32 KB
  float*  gsc   = (float*) (ws + 6324224);    // 256 KB
  int*    topki = (int*)   (ws + 6586368);    // 3 KB
  float*  uselw = (float*) (ws + 6589440);    // 3 KB
  float*  qproj = (float*) (ws + 6592512);    // 128 KB
  float*  kproj = (float*) (ws + 6723584);    // 3 MB
  float*  vproj = (float*) (ws + 9869312);    // 3 MB

  float* out   = (float*)d_out;
  float* fused = out;                 // (B, D)      32768
  float* uavg  = out + 32768;         // (B, 1)      32
  float* etok  = out + 32800;         // (B, L)      65536
  float* attn  = out + 98336;         // (B,H,1,24)  6144

  k_prep  <<<1024, 256, 0, stream>>>(gW1, Wk, Wv, gW1bf, Wkbf, Wvbf);
  k_c0    <<<BB,   256, 0, stream>>>(q_vec, gW1, c0);
  k_qproj <<<BB,   256, 0, stream>>>(q_vec, Wq, bq, qproj);
  k_gate  <<<BB * (LL / 16), 256, 0, stream>>>(q_vec, kv, gW1bf, c0, gb1, gW2, gb2, etok, gsc);
  k_topk  <<<BB,   256, 0, stream>>>(gsc, mask, topki, uselw, uavg);
  k_kvproj<<<BB,   256, 0, stream>>>(kv, topki, Wkbf, Wvbf, bk, bv, kproj, vproj);
  k_attn  <<<BB,   256, 0, stream>>>(qproj, kproj, vproj, uselw, q_vec, Wo, bo,
                                     ln_g, ln_b, fused, attn);
}
